// TopoGraphGNN_75746043232715
// MI455X (gfx1250) — matrix operand order repaired and checked by reading, not compiled, not run on hardware
//
#include <hip/hip_runtime.h>
#include <hip/hip_bf16.h>

// ---------------- model constants ----------------
#define Bx 64
#define Tt 512
#define Lx 16
#define FN 32
#define FE 16
#define Hx 128
#define NE 15           // edges = L-1
#define H3 (3*Hx)

// packed bf16 weight offsets in d_ws (elements)
#define OFF_NW0   0
#define OFF_MW0   4096
#define OFF_WIH0  20480
#define OFF_WHH0  69632
#define OFF_NW1   118784
#define OFF_MW1   135168
#define OFF_WIH1  151552
#define OFF_WHH1  200704
#define WS_ELEMS  249856

// d_in indices (pytree: dicts flattened by sorted key)
// 0 node_seq, 1 edge_seq,
// layer0: 2 b_hh 3 b_ih 4 e_b1 5 e_b2 6 e_w1 7 e_w2 8 m_b 9 m_w 10 n_b 11 n_w 12 w_hh 13 w_ih
// layer1: 14..25 same order
// link_healthy_head: 26 b1 27 b2 28 w1 29 w2
// link_node_head:    30 b1 31 b2 32 w1 33 w2
// motor_cond_head:   34 b1 35 b2 36 w1 37 w2
// 38 motor_healthy_bias

typedef __attribute__((ext_vector_type(16))) __bf16 v16bf;
typedef __attribute__((ext_vector_type(8)))  float  v8f;
typedef __attribute__((ext_vector_type(4)))  unsigned int u32x4;

union Frag {
  v16bf v;
  u32x4 q[2];
};

__device__ __forceinline__ unsigned short f2bf(float f) {
  unsigned int u = __float_as_uint(f);
  u += 0x7FFFu + ((u >> 16) & 1u);       // round to nearest even
  return (unsigned short)(u >> 16);
}
__device__ __forceinline__ float sigm(float x) { return 1.0f / (1.0f + __expf(-x)); }

// ---------------- weight pre-pack: fp32 row-major (K,N) -> bf16 WMMA B-fragments ----
// fragment layout: tile (kt,nt): 32 lanes x 16 bf16 contiguous.
// lane l<16  : N = nt*16+l,      K = kt*32 + 0..15
// lane l>=16 : N = nt*16+(l-16), K = kt*32 + 16..31
__global__ void pack_w(const float* __restrict__ W, unsigned short* __restrict__ dst,
                       int K, int N) {
  int idx = blockIdx.x * blockDim.x + threadIdx.x;
  if (idx >= K * N) return;
  int k = idx / N, n = idx % N;
  int kt = k >> 5, kin = k & 31, nt = n >> 4;
  int lane = ((kin & 16) ? 16 : 0) + (n & 15);
  int j = kin & 15;
  int ntiles = N >> 4;
  dst[((((kt * ntiles + nt) * 32) + lane) << 4) + j] = f2bf(W[idx]);
}

// A fragment (16x32 bf16 tile) from LDS row-major buffer
__device__ __forceinline__ void load_afrag(Frag& a, const unsigned short* Abf, int lda,
                                           int k0, int lane) {
  int row = lane & 15;
  int kh  = (lane >> 4) << 4;            // lanes 0-15: K k0+0..15; 16-31: k0+16..31
  const u32x4* p0 = (const u32x4*)(Abf + row * lda + k0 + kh);
  const u32x4* p1 = (const u32x4*)(Abf + row * lda + k0 + 8 + kh);
  a.q[0] = p0[0];
  a.q[1] = p1[0];
}
// B fragment (32x16 bf16 tile) from packed global weights
__device__ __forceinline__ void load_bfrag(Frag& b, const unsigned short* __restrict__ Bws,
                                           int ntiles, int ki, int nt, int lane) {
  const u32x4* p = (const u32x4*)(Bws + ((((ki * ntiles + nt) * 32) + lane) << 4));
  b.q[0] = p[0];
  b.q[1] = p[1];
}

struct KParams {
  const float* in[39];
  const unsigned short* wbf;
  float* out;
};

__global__ __launch_bounds__(128) void gnn_scan_kernel(KParams p) {
  const int b    = blockIdx.x;
  const int tid  = threadIdx.x;
  const int lane = tid & 31;
  const int wid  = tid >> 5;

  // ---------------- LDS state (~78 KB) ----------------
  __shared__ float          h_s [Lx * Hx];
  __shared__ unsigned short hbf_s[Lx * Hx];
  __shared__ float          zf_s[Lx * Hx];
  __shared__ unsigned short zbf_s[Lx * Hx];
  __shared__ unsigned short mbf_s[Lx * Hx];
  __shared__ unsigned short xbf_s[Lx * FN];
  __shared__ float          gi_s[Lx * H3];
  __shared__ float          gh_s[Lx * H3];
  __shared__ float          et_s[NE * FE];
  __shared__ float          we_s[16];
  float* ehid = gi_s;                     // alias: edge-MLP hidden (1920 <= 6144)

  const float* node_seq = p.in[0];
  const float* edge_seq = p.in[1];

  for (int idx = tid; idx < Lx * Hx; idx += 128) h_s[idx] = 0.0f;
  __syncthreads();

  // ================= time scan =================
  #pragma unroll 1
  for (int t = 0; t < Tt; ++t) {
    const float* xp = node_seq + (size_t)(b * Tt + t) * (Lx * FN);
    const float* ep = edge_seq + (size_t)(b * Tt + t) * (NE * FE);
    for (int idx = tid; idx < Lx * FN; idx += 128) xbf_s[idx] = f2bf(xp[idx]);
    for (int idx = tid; idx < NE * FE; idx += 128) et_s[idx] = ep[idx];
    if (t + 1 < Tt) {     // prefetch next step (global_prefetch_b8)
      if (tid < 32) __builtin_prefetch(xp + Lx * FN + tid * 16, 0, 0);
      if (tid < 15) __builtin_prefetch(ep + NE * FE + tid * 16, 0, 0);
    }
    __syncthreads();

    // fully unrolled: layer is compile-time, all weight pointers fold to
    // p.wbf + CONST -> provably-global loads (no FLAT, no per-tile branches)
    #pragma unroll
    for (int layer = 0; layer < 2; ++layer) {
      const float* eb1 = (layer == 0) ? p.in[4]  : p.in[16];
      const float* eb2 = (layer == 0) ? p.in[5]  : p.in[17];
      const float* ew1 = (layer == 0) ? p.in[6]  : p.in[18];
      const float* ew2 = (layer == 0) ? p.in[7]  : p.in[19];
      const float* mb  = (layer == 0) ? p.in[8]  : p.in[20];
      const float* nb  = (layer == 0) ? p.in[10] : p.in[22];
      const float* bih = (layer == 0) ? p.in[3]  : p.in[15];
      const float* bhh = (layer == 0) ? p.in[2]  : p.in[14];
      const unsigned short* __restrict__ wnw = p.wbf + ((layer == 0) ? OFF_NW0  : OFF_NW1);
      const unsigned short* __restrict__ wmw = p.wbf + ((layer == 0) ? OFF_MW0  : OFF_MW1);
      const unsigned short* __restrict__ wih = p.wbf + ((layer == 0) ? OFF_WIH0 : OFF_WIH1);
      const unsigned short* __restrict__ whh = p.wbf + ((layer == 0) ? OFF_WHH0 : OFF_WHH1);

      // (a) edge MLP hidden: relu(e @ e_w1 + e_b1) -- thread = hidden unit j
      {
        float bj = eb1[tid];
        for (int e = 0; e < NE; ++e) {
          float s = bj;
          #pragma unroll
          for (int k = 0; k < FE; ++k) s += et_s[e * FE + k] * ew1[k * Hx + tid];
          ehid[e * Hx + tid] = fmaxf(s, 0.0f);
        }
      }
      __syncthreads();
      if (tid < NE) {     // w_e = sigmoid(hid @ e_w2 + e_b2)
        float s = eb2[0];
        for (int j = 0; j < Hx; ++j) s += ehid[tid * Hx + j] * ew2[j];
        we_s[tid] = sigm(s);
      }
      __syncthreads();

      // (b) message passing + bf16 copies of m and h (WMMA A-operands)
      for (int idx = tid; idx < Lx * Hx; idx += 128) {
        int l = idx >> 7;
        float wl = (l > 0)      ? we_s[l - 1] : 0.0f;
        float wr = (l < Lx - 1) ? we_s[l]     : 0.0f;
        float hl = (l > 0)      ? h_s[idx - Hx] : 0.0f;
        float hr = (l < Lx - 1) ? h_s[idx + Hx] : 0.0f;
        float hc = h_s[idx];
        float deg = fmaxf(wl + wr + 1.0f, 1e-6f);
        mbf_s[idx] = f2bf((wl * hl + wr * hr + hc) / deg);
        hbf_s[idx] = f2bf(hc);
      }
      __syncthreads();

      // (c) z = node_x @ n_w + m @ m_w + n_b + m_b   (16x128, 8 N-tiles)
      {
        Frag am[4], an[4];
        #pragma unroll
        for (int ki = 0; ki < 4; ++ki) load_afrag(am[ki], mbf_s, Hx, ki * 32, lane);
        if (layer == 0) load_afrag(an[0], xbf_s, FN, 0, lane);
        else {
          #pragma unroll
          for (int ki = 0; ki < 4; ++ki) load_afrag(an[ki], hbf_s, Hx, ki * 32, lane);
        }
        for (int nt = wid; nt < 8; nt += 4) {
          v8f acc = {};
          #pragma unroll
          for (int ki = 0; ki < 4; ++ki) {
            Frag bf;
            load_bfrag(bf, wmw, 8, ki, nt, lane);
            acc = __builtin_amdgcn_wmma_f32_16x16x32_bf16(false, am[ki].v, false, bf.v,
                                                          (short)0, acc, false, false);
          }
          if (layer == 0) {
            Frag bf;
            load_bfrag(bf, wnw, 8, 0, nt, lane);
            acc = __builtin_amdgcn_wmma_f32_16x16x32_bf16(false, an[0].v, false, bf.v,
                                                          (short)0, acc, false, false);
          } else {
            #pragma unroll
            for (int ki = 0; ki < 4; ++ki) {
              Frag bf;
              load_bfrag(bf, wnw, 8, ki, nt, lane);
              acc = __builtin_amdgcn_wmma_f32_16x16x32_bf16(false, an[ki].v, false, bf.v,
                                                            (short)0, acc, false, false);
            }
          }
          int col = (nt << 4) + (lane & 15);
          int rb  = (lane >> 4) << 3;
          float bias = nb[col] + mb[col];
          #pragma unroll
          for (int r = 0; r < 8; ++r) {
            float v = acc[r] + bias;
            zf_s [(rb + r) * Hx + col] = v;
            zbf_s[(rb + r) * Hx + col] = f2bf(v);
          }
        }
      }
      __syncthreads();

      // (d) gi = z @ w_ih + b_ih     (16x384, 24 N-tiles)
      {
        Frag az[4];
        #pragma unroll
        for (int ki = 0; ki < 4; ++ki) load_afrag(az[ki], zbf_s, Hx, ki * 32, lane);
        for (int nt = wid; nt < 24; nt += 4) {
          v8f acc = {};
          #pragma unroll
          for (int ki = 0; ki < 4; ++ki) {
            Frag bf;
            load_bfrag(bf, wih, 24, ki, nt, lane);
            acc = __builtin_amdgcn_wmma_f32_16x16x32_bf16(false, az[ki].v, false, bf.v,
                                                          (short)0, acc, false, false);
          }
          int col = (nt << 4) + (lane & 15);
          int rb  = (lane >> 4) << 3;
          float bias = bih[col];
          #pragma unroll
          for (int r = 0; r < 8; ++r) gi_s[(rb + r) * H3 + col] = acc[r] + bias;
        }
      }
      // (e) gh = h_prev @ w_hh + b_hh
      {
        Frag ah[4];
        #pragma unroll
        for (int ki = 0; ki < 4; ++ki) load_afrag(ah[ki], hbf_s, Hx, ki * 32, lane);
        for (int nt = wid; nt < 24; nt += 4) {
          v8f acc = {};
          #pragma unroll
          for (int ki = 0; ki < 4; ++ki) {
            Frag bf;
            load_bfrag(bf, whh, 24, ki, nt, lane);
            acc = __builtin_amdgcn_wmma_f32_16x16x32_bf16(false, ah[ki].v, false, bf.v,
                                                          (short)0, acc, false, false);
          }
          int col = (nt << 4) + (lane & 15);
          int rb  = (lane >> 4) << 3;
          float bias = bhh[col];
          #pragma unroll
          for (int r = 0; r < 8; ++r) gh_s[(rb + r) * H3 + col] = acc[r] + bias;
        }
      }
      __syncthreads();

      // (f) GRU gating (torch order r,z,n), in-place h update
      for (int idx = tid; idx < Lx * Hx; idx += 128) {
        int l = idx >> 7, j = idx & 127;
        int base = l * H3;
        float r  = sigm(gi_s[base + j]          + gh_s[base + j]);
        float zg = sigm(gi_s[base + Hx + j]     + gh_s[base + Hx + j]);
        float n  = tanhf(gi_s[base + 2 * Hx + j] + r * gh_s[base + 2 * Hx + j]);
        h_s[idx] = (1.0f - zg) * n + zg * h_s[idx];
      }
      __syncthreads();
    }
  }

  // ================= heads (once) =================
  // link_node_head hidden -> gi_s
  {
    const float* W1 = p.in[32];
    float bj = p.in[30][tid];
    for (int l = 0; l < Lx; ++l) {
      float s = bj;
      for (int k = 0; k < Hx; ++k) s += h_s[l * Hx + k] * W1[k * Hx + tid];
      gi_s[l * Hx + tid] = fmaxf(s, 0.0f);
    }
  }
  __syncthreads();
  if (tid < Lx) {                       // node scores -> we_s
    const float* W2 = p.in[33];
    float s = p.in[31][0];
    for (int j = 0; j < Hx; ++j) s += gi_s[tid * Hx + j] * W2[j];
    we_s[tid] = s;
  }
  if (tid < Hx) {                       // pooled -> zf_s[0..127]
    float s = 0.0f;
    for (int l = 0; l < Lx; ++l) s += h_s[l * Hx + tid];
    zf_s[tid] = s * (1.0f / Lx);
  }
  __syncthreads();
  {                                     // link_healthy_head hidden -> gh_s[0..127]
    const float* W1 = p.in[28];
    float s = p.in[26][tid];
    for (int k = 0; k < Hx; ++k) s += zf_s[k] * W1[k * Hx + tid];
    gh_s[tid] = fmaxf(s, 0.0f);
  }
  __syncthreads();
  if (tid == 0) {
    const float* W2 = p.in[29];
    float s = p.in[27][0];
    for (int j = 0; j < Hx; ++j) s += gh_s[j] * W2[j];
    zf_s[Hx] = s;                       // healthy
  }
  __syncthreads();
  float healthy = zf_s[Hx];
  {                                     // motor_cond_head hidden -> gh_s (2048 floats)
    const float* W1 = p.in[36];
    float bj = p.in[34][tid];
    for (int l = 0; l < Lx; ++l) {
      float s = bj;
      for (int k = 0; k < Hx; ++k) s += h_s[l * Hx + k] * W1[k * Hx + tid];
      gh_s[l * Hx + tid] = fmaxf(s, 0.0f);
    }
  }
  __syncthreads();

  float* link = p.out;                  // (B, 17)
  float* mot  = p.out + Bx * 17;        // (B, 129)
  if (tid == 0) link[b * 17] = healthy;
  if (tid < Lx) link[b * 17 + 1 + tid] = we_s[tid];
  if (tid == 0) mot[b * 129] = healthy + p.in[38][0];
  {
    int l = tid >> 3, mm = tid & 7;     // 128 threads = 16x8 outputs
    const float* W2 = p.in[37];
    float s = p.in[35][mm];
    for (int j = 0; j < Hx; ++j) s += gh_s[l * Hx + j] * W2[j * 8 + mm];
    mot[b * 129 + 1 + tid] = s + we_s[l];
  }
}

extern "C" void kernel_launch(void* const* d_in, const int* in_sizes, int n_in,
                              void* d_out, int out_size, void* d_ws, size_t ws_size,
                              hipStream_t stream) {
  (void)in_sizes; (void)n_in; (void)out_size; (void)ws_size;
  unsigned short* wbf = (unsigned short*)d_ws;

  struct { const float* W; int K; int N; int off; } mats[8] = {
    { (const float*)d_in[11],  32, 128, OFF_NW0  },
    { (const float*)d_in[9],  128, 128, OFF_MW0  },
    { (const float*)d_in[13], 128, 384, OFF_WIH0 },
    { (const float*)d_in[12], 128, 384, OFF_WHH0 },
    { (const float*)d_in[23], 128, 128, OFF_NW1  },
    { (const float*)d_in[21], 128, 128, OFF_MW1  },
    { (const float*)d_in[25], 128, 384, OFF_WIH1 },
    { (const float*)d_in[24], 128, 384, OFF_WHH1 },
  };
  for (int i = 0; i < 8; ++i) {
    int total = mats[i].K * mats[i].N;
    pack_w<<<(total + 255) / 256, 256, 0, stream>>>(mats[i].W, wbf + mats[i].off,
                                                    mats[i].K, mats[i].N);
  }

  KParams p;
  for (int i = 0; i < 39; ++i) p.in[i] = (const float*)d_in[i];
  p.wbf = wbf;
  p.out = (float*)d_out;
  gnn_scan_kernel<<<Bx, 128, 0, stream>>>(p);
}
